// Classifier_72146860638958
// MI455X (gfx1250) — matrix operand order repaired and checked by reading, not compiled, not run on hardware
//
#include <hip/hip_runtime.h>
#include <stdint.h>

#define B_   32
#define T_   256
#define DIN_ 768
#define DH_  512
#define DE_  25
#define C_   9
#define K3DIN 2304       // 3*DIN
#define H3    1536       // 3*DH

// CDNA5 async global->LDS copies via inline asm (no clang builtin confirmed).
// Flip to 0 if the assembler rejects the mnemonics.
#define USE_ASYNC_LDS 1

typedef __attribute__((ext_vector_type(16))) __bf16 v16bf;
typedef __attribute__((ext_vector_type(8)))  float  v8f;

union FragBF { uint4 q[2]; v16bf v; };

__device__ __forceinline__ unsigned short f2bf(float f) {
    union { float f; unsigned u; } x; x.f = f;
    unsigned r = 0x7FFFu + ((x.u >> 16) & 1u);   // round-to-nearest-even
    x.u += r;
    return (unsigned short)(x.u >> 16);
}
__device__ __forceinline__ float bf2f(unsigned short h) {
    union { unsigned u; float f; } x; x.u = ((unsigned)h) << 16;
    return x.f;
}

#if USE_ASYNC_LDS
// One 16-byte async DMA: LDS[lds_off + io] = MEM[gaddr + io], ASYNCcnt++
__device__ __forceinline__ void async_b128(unsigned lds_off, const void* gaddr) {
    asm volatile("global_load_async_to_lds_b128 %0, %1, off"
                 :: "v"(lds_off), "v"((unsigned long long)(uintptr_t)gaddr)
                 : "memory");
}
__device__ __forceinline__ void async_b128_o32(unsigned lds_off, const void* gaddr) {
    asm volatile("global_load_async_to_lds_b128 %0, %1, off offset:16"
                 :: "v"(lds_off), "v"((unsigned long long)(uintptr_t)gaddr)
                 : "memory");
}
__device__ __forceinline__ void wait_async0() {
    asm volatile("s_wait_asynccnt 0" ::: "memory");
}
// low 32 bits of a generic pointer to __shared__ == LDS byte offset (ISA 10.2)
__device__ __forceinline__ unsigned lds_off(const void* p) {
    return (unsigned)(uintptr_t)p;
}
#endif

// ---------------------------------------------------------------------------
// Phase 0: fp32 -> bf16 conversion of X (with pad rows), Wih, Whh; zero h
// state ping-pong buffers and barrier counters (deterministic every launch).
// ---------------------------------------------------------------------------
__global__ void prep_kernel(const float* __restrict__ X, const float* __restrict__ pad,
                            const float* __restrict__ WihF, const float* __restrict__ WihB,
                            const float* __restrict__ WhhF, const float* __restrict__ WhhB,
                            unsigned short* xpad, unsigned short* wihF, unsigned short* wihB,
                            unsigned short* whhF, unsigned short* whhB,
                            unsigned short* hbuf, int* cnt)
{
    long long idx = (long long)blockIdx.x * blockDim.x + threadIdx.x;
    const long long N0 = (long long)B_ * (T_ + 2) * DIN_;
    const long long N1 = (long long)H3 * K3DIN;
    const long long N2 = (long long)H3 * DH_;
    const long long NH = 4LL * B_ * DH_;   // 2 dirs x 2 ping-pong x [32][512]
    const long long NC = 2 * 256;
    if (idx < N0) {
        int k  = (int)(idx % DIN_);
        long long bt = idx / DIN_;
        int tt = (int)(bt % (T_ + 2));
        int b  = (int)(bt / (T_ + 2));
        float v = (tt == 0 || tt == T_ + 1) ? pad[k]
                                            : X[((long long)b * T_ + (tt - 1)) * DIN_ + k];
        xpad[idx] = f2bf(v); return;
    }
    idx -= N0;
    if (idx < N1) { wihF[idx] = f2bf(WihF[idx]); return; }
    idx -= N1;
    if (idx < N1) { wihB[idx] = f2bf(WihB[idx]); return; }
    idx -= N1;
    if (idx < N2) { whhF[idx] = f2bf(WhhF[idx]); return; }
    idx -= N2;
    if (idx < N2) { whhB[idx] = f2bf(WhhB[idx]); return; }
    idx -= N2;
    if (idx < NH) { hbuf[idx] = 0; return; }
    idx -= NH;
    if (idx < NC) { cnt[idx] = 0; return; }
}

// ---------------------------------------------------------------------------
// Phase 1: GI[dir] = lmr @ Wih^T + bih   (M=8192, N=1536, K=2304, bf16 WMMA)
// Tile: 128x128 per WG, 8 waves in 2(M)x4(N); wave = 64x32 = 8 accum tiles.
// Double-buffered LDS with async global->LDS DMA: one s_barrier per k-chunk,
// next tile's DMA overlaps the current tile's WMMA work.
// ---------------------------------------------------------------------------
__global__ __launch_bounds__(256) void igemm_kernel(
    const unsigned short* __restrict__ xpad,   // [B][T+2][DIN] bf16
    const unsigned short* __restrict__ wihF,   // [1536][2304] bf16 (N-major,K-contig)
    const unsigned short* __restrict__ wihB,
    const float* __restrict__ bihF, const float* __restrict__ bihB,
    float* __restrict__ giF, float* __restrict__ giB)
{
    __shared__ __align__(16) unsigned short As[2][128 * 40];  // 128x32 padded to 40
    __shared__ __align__(16) unsigned short Bs[2][128 * 40];

    const int dir = blockIdx.y;
    const unsigned short* W = dir ? wihB : wihF;
    const float* bih = dir ? bihB : bihF;
    float* gi = dir ? giB : giF;

    const int mt = blockIdx.x / 12;
    const int nt = blockIdx.x % 12;
    const int tid  = threadIdx.x;
    const int lane = tid & 31, w = tid >> 5;
    const int lr = lane & 15, lh = lane >> 4;
    const int wm = w >> 2, wn = w & 3;

    v8f acc[4][2];
#pragma unroll
    for (int i = 0; i < 4; i++)
#pragma unroll
        for (int j = 0; j < 2; j++) acc[i][j] = (v8f){0,0,0,0,0,0,0,0};

    const int lrow  = tid >> 1;   // 0..127  (tile row this thread stages)
    const int lhalf = tid & 1;    // which 16-bf16 half of the 32-wide k-chunk
    const int m  = mt * 128 + lrow;
    const int ab = m >> 8, at = m & 255;
    const int nrow = nt * 128 + lrow;

    // stage tile kc into buffer kc&1 (2x b128 per thread for A, 2x for B)
    auto issue_tile = [&](int kc) {
        const int ks = kc * 32;
        const int buf = kc & 1;
        const int srcT = at + ks / DIN_;
        const int srcK = ks % DIN_;
        const unsigned short* ga = xpad +
            (((long long)ab * (T_ + 2) + srcT) * DIN_ + srcK + lhalf * 16);
        const unsigned short* gb = W + ((long long)nrow * K3DIN + ks + lhalf * 16);
#if USE_ASYNC_LDS
        unsigned la = lds_off(&As[buf][lrow * 40 + lhalf * 16]);
        unsigned lb = lds_off(&Bs[buf][lrow * 40 + lhalf * 16]);
        async_b128(la, ga);  async_b128_o32(la, ga);     // +16B on both sides
        async_b128(lb, gb);  async_b128_o32(lb, gb);
#else
        uint4* sa = (uint4*)(&As[buf][lrow * 40 + lhalf * 16]);
        uint4* sb = (uint4*)(&Bs[buf][lrow * 40 + lhalf * 16]);
        sa[0] = ((const uint4*)ga)[0]; sa[1] = ((const uint4*)ga)[1];
        sb[0] = ((const uint4*)gb)[0]; sb[1] = ((const uint4*)gb)[1];
#endif
    };

    issue_tile(0);
    for (int kc = 0; kc < 72; ++kc) {
        const int buf = kc & 1;
#if USE_ASYNC_LDS
        wait_async0();           // my DMA for tile kc has landed
#endif
        __syncthreads();         // everyone's tile kc landed; tile kc-1 fully read
        if (kc + 1 < 72) issue_tile(kc + 1);   // DMA overlaps WMMA below

        FragBF af[4], bfr[2];
#pragma unroll
        for (int i = 0; i < 4; i++) {
            int row = wm * 64 + i * 16 + lr;
            af[i].q[0] = *(const uint4*)(&As[buf][row * 40 + lh * 8]);
            af[i].q[1] = *(const uint4*)(&As[buf][row * 40 + 16 + lh * 8]);
        }
#pragma unroll
        for (int j = 0; j < 2; j++) {
            int col = wn * 32 + j * 16 + lr;
            bfr[j].q[0] = *(const uint4*)(&Bs[buf][col * 40 + lh * 8]);
            bfr[j].q[1] = *(const uint4*)(&Bs[buf][col * 40 + 16 + lh * 8]);
        }
#pragma unroll
        for (int i = 0; i < 4; i++)
#pragma unroll
            for (int j = 0; j < 2; j++)
                acc[i][j] = __builtin_amdgcn_wmma_f32_16x16x32_bf16(
                    false, af[i].v, false, bfr[j].v, (short)0, acc[i][j], false, false);
    }

    // epilogue: + bih, store fp32 GI
#pragma unroll
    for (int i = 0; i < 4; i++) {
#pragma unroll
        for (int j = 0; j < 2; j++) {
            int colg = nt * 128 + wn * 32 + j * 16 + lr;
            float bv = bih[colg];
#pragma unroll
            for (int v = 0; v < 8; v++) {
                int rowg = mt * 128 + wm * 64 + i * 16 + lh * 8 + v;
                gi[(long long)rowg * H3 + colg] = acc[i][j][v] + bv;
            }
        }
    }
}

// ---------------------------------------------------------------------------
// Phase 2: GRU recurrence. 16 WGs per direction; WG g owns hidden units
// [32g, 32g+32). Its Whh slice (96 rows x 512) lives in LDS (bf16). Each
// step: async-stage h_prev (bf16) to LDS, 12 waves do gh = h @ WhhSlice^T via
// WMMA, prefetch next step's gi, elementwise gates, write h slice + output_h,
// then per-step global barrier across the direction's 16 WGs.
// ---------------------------------------------------------------------------
__global__ __launch_bounds__(384) void gru_step_kernel(
    const float* __restrict__ giF, const float* __restrict__ giB,
    const unsigned short* __restrict__ whhF, const unsigned short* __restrict__ whhB,
    const float* __restrict__ bhhF, const float* __restrict__ bhhB,
    unsigned short* __restrict__ hbuf,   // [dir][pp][32][512] bf16
    int* __restrict__ cnt,               // [dir][256]
    float* __restrict__ outh)            // [B][T][1024]
{
    extern __shared__ __align__(16) char smem_raw[];
    unsigned short* WhhS = (unsigned short*)smem_raw;   // 96 x 520 (padded)
    unsigned short* hS   = WhhS + 96 * 520;             // 32 x 520 (padded)
    float*          ghS  = (float*)(hS + 32 * 520);     // 32 x 100 (padded)

    const int dir = blockIdx.x >> 4;
    const int g   = blockIdx.x & 15;
    const int j0  = g * 32;
    const int tid = threadIdx.x;
    const int lane = tid & 31, w = tid >> 5;       // 12 waves
    const int lr = lane & 15, lh = lane >> 4;
    const int mi = w / 6, ni = w % 6;              // 2(M) x 6(N) tiles

    const float* gi = dir ? giB : giF;
    const unsigned short* Whh = dir ? whhB : whhF;
    const float* bhh = dir ? bhhB : bhhF;
    int* mycnt = cnt + dir * 256;
    unsigned short* h0 = hbuf + dir * 2 * (B_ * DH_);

    // preload Whh slice rows { j0.., 512+j0.., 1024+j0.. } (96 rows x 512 bf16)
    for (int i = tid; i < 96 * 64; i += 384) {         // 64 uint4 per row
        int row = i >> 6, q = i & 63;
        int grow = (row >> 5) * DH_ + j0 + (row & 31);
        *(uint4*)(WhhS + row * 520 + q * 8) =
            *(const uint4*)(Whh + (long long)grow * DH_ + q * 8);
    }
    __syncthreads();

    for (int s = 0; s < T_; ++s) {
        const int t = dir ? (T_ - 1 - s) : s;
        const unsigned short* hprev = h0 + (s & 1) * (B_ * DH_);
        unsigned short* hnext = h0 + (((s & 1) ^ 1)) * (B_ * DH_);

        // stage h_prev -> LDS (32 x 512 bf16)
        for (int i = tid; i < 32 * 64; i += 384) {
            int row = i >> 6, q = i & 63;
#if USE_ASYNC_LDS
            async_b128(lds_off(hS + row * 520 + q * 8), hprev + row * DH_ + q * 8);
#else
            *(uint4*)(hS + row * 520 + q * 8) =
                *(const uint4*)(hprev + row * DH_ + q * 8);
#endif
        }
#if USE_ASYNC_LDS
        wait_async0();
#endif
        __syncthreads();

        v8f acc = (v8f){0,0,0,0,0,0,0,0};
#pragma unroll
        for (int kc = 0; kc < 16; ++kc) {
            FragBF a, b;
            int mrow = mi * 16 + lr;
            a.q[0] = *(const uint4*)(hS + mrow * 520 + kc * 32 + lh * 8);
            a.q[1] = *(const uint4*)(hS + mrow * 520 + kc * 32 + 16 + lh * 8);
            int ncol = ni * 16 + lr;
            b.q[0] = *(const uint4*)(WhhS + ncol * 520 + kc * 32 + lh * 8);
            b.q[1] = *(const uint4*)(WhhS + ncol * 520 + kc * 32 + 16 + lh * 8);
            acc = __builtin_amdgcn_wmma_f32_16x16x32_bf16(
                false, a.v, false, b.v, (short)0, acc, false, false);
        }
#pragma unroll
        for (int v = 0; v < 8; v++)
            ghS[(mi * 16 + lh * 8 + v) * 100 + ni * 16 + lr] = acc[v];

        // near-cache prefetch of next step's gi slice while gates compute
        {
            int tn = dir ? (t - 1) : (t + 1);
            if (0 <= tn && tn < T_) {
                for (int idx = tid; idx < B_ * 3; idx += 384) {
                    int b = idx / 3, r3 = idx % 3;
                    __builtin_prefetch(gi + (long long)(b * T_ + tn) * H3 + r3 * DH_ + j0, 0, 3);
                }
            }
        }
        __syncthreads();

        // elementwise gates for the 32x32 (batch, hidden) slice
        for (int idx = tid; idx < B_ * 32; idx += 384) {
            int b  = idx >> 5, j = idx & 31, jg = j0 + j;
            const float* gir = gi + (long long)(b * T_ + t) * H3;
            float ir = gir[jg], iz = gir[DH_ + jg], in_ = gir[2 * DH_ + jg];
            float hr = ghS[b * 100 + j]        + bhh[jg];
            float hz = ghS[b * 100 + 32 + j]   + bhh[DH_ + jg];
            float hn = ghS[b * 100 + 64 + j]   + bhh[2 * DH_ + jg];
            float hp = bf2f(hS[b * 520 + jg]);
            float r = 1.f / (1.f + __expf(-(ir + hr)));
            float z = 1.f / (1.f + __expf(-(iz + hz)));
            float e2 = __expf(2.f * (in_ + r * hn));
            float n  = (e2 - 1.f) / (e2 + 1.f);              // tanh
            float h  = (1.f - z) * n + z * hp;
            outh[(long long)(b * T_ + t) * (2 * DH_) + dir * DH_ + jg] = h;
            hnext[b * DH_ + jg] = f2bf(h);
        }

        // per-step barrier across the 16 WGs of this direction
        __threadfence();
        __syncthreads();
        if (tid == 0) {
            atomicAdd(mycnt + s, 1);
            while (atomicAdd(mycnt + s, 0) < 16) { __builtin_amdgcn_s_sleep(1); }
        }
        __syncthreads();
        __threadfence();
    }
}

// ---------------------------------------------------------------------------
// Phase 3: sequential decoder. One wave per batch row; Wc cached in LDS;
// wave32 shfl_xor butterfly reductions; argmax feeds next step's label emb.
// ---------------------------------------------------------------------------
__global__ __launch_bounds__(32) void decode_kernel(
    const float* __restrict__ outh,      // [B][T][1024]
    const float* __restrict__ Wc,        // [9][1049]
    const float* __restrict__ bc,        // [9]
    const float* __restrict__ emb,       // [10][25]
    float* __restrict__ outc)            // [B*T][9]
{
    __shared__ float WcS[C_ * 1056];
    __shared__ float embS[10 * DE_];
    __shared__ float bcS[C_];
    const int b = blockIdx.x;
    const int lane = threadIdx.x;

    for (int i = lane; i < C_ * 1049; i += 32)
        WcS[(i / 1049) * 1056 + (i % 1049)] = Wc[i];
    for (int i = lane; i < 10 * DE_; i += 32) embS[i] = emb[i];
    if (lane < C_) bcS[lane] = bc[lane];
    __syncthreads();

    int prev = C_;   // start token = last row of label_emb
    for (int t = 0; t < T_; ++t) {
        const float* ht = outh + (long long)(b * T_ + t) * (2 * DH_);
        float hreg[32];
#pragma unroll
        for (int u = 0; u < 32; ++u) hreg[u] = ht[lane + 32 * u];
        float lab = (lane < DE_) ? embS[prev * DE_ + lane] : 0.f;

        float lg[C_];
#pragma unroll
        for (int c = 0; c < C_; ++c) {
            const float* wrow = WcS + c * 1056;
            float sum = 0.f;
#pragma unroll
            for (int u = 0; u < 32; ++u) sum += hreg[u] * wrow[lane + 32 * u];
            sum += lab * ((lane < DE_) ? wrow[1024 + lane] : 0.f);
#pragma unroll
            for (int off = 16; off; off >>= 1) sum += __shfl_xor(sum, off, 32);
            lg[c] = sum + bcS[c];
        }
        float mx = lg[0]; int am = 0;
#pragma unroll
        for (int c = 1; c < C_; ++c) if (lg[c] > mx) { mx = lg[c]; am = c; }
        float den = 0.f;
#pragma unroll
        for (int c = 0; c < C_; ++c) { lg[c] = __expf(lg[c] - mx); den += lg[c]; }
        float inv = 1.f / den;
        if (lane < C_) outc[(long long)(b * T_ + t) * C_ + lane] = lg[lane] * inv;
        prev = am;
    }
}

// ---------------------------------------------------------------------------
extern "C" void kernel_launch(void* const* d_in, const int* in_sizes, int n_in,
                              void* d_out, int out_size, void* d_ws, size_t ws_size,
                              hipStream_t stream)
{
    const float* X    = (const float*)d_in[0];
    const float* pad  = (const float*)d_in[1];
    const float* emb  = (const float*)d_in[2];
    const float* WihF = (const float*)d_in[3];
    const float* WhhF = (const float*)d_in[4];
    const float* bihF = (const float*)d_in[5];
    const float* bhhF = (const float*)d_in[6];
    const float* WihB = (const float*)d_in[7];
    const float* WhhB = (const float*)d_in[8];
    const float* bihB = (const float*)d_in[9];
    const float* bhhB = (const float*)d_in[10];
    const float* Wc   = (const float*)d_in[11];
    const float* bc   = (const float*)d_in[12];

    float* outh = (float*)d_out;                          // [32][256][1024]
    float* outc = outh + (size_t)B_ * T_ * 2 * DH_;       // [8192][9]

    char* ws = (char*)d_ws;
    unsigned short* xpad = (unsigned short*)(ws + 0);             // 12,681,216 B
    unsigned short* wihF = (unsigned short*)(ws + 12681216);      //  7,077,888 B
    unsigned short* wihB = (unsigned short*)(ws + 19759104);      //  7,077,888 B
    unsigned short* whhF = (unsigned short*)(ws + 26836992);      //  1,572,864 B
    unsigned short* whhB = (unsigned short*)(ws + 28409856);      //  1,572,864 B
    float*          giF  = (float*)(ws + 29982720);               // 50,331,648 B
    float*          giB  = (float*)(ws + 80314368);               // 50,331,648 B
    unsigned short* hbuf = (unsigned short*)(ws + 130646016);     //    131,072 B
    int*            cnt  = (int*)(ws + 130777088);                //      2,048 B

    // Phase 0: conversions + state/counter init (deterministic per launch)
    {
        long long total = (long long)B_ * (T_ + 2) * DIN_
                        + 2LL * H3 * K3DIN + 2LL * H3 * DH_
                        + 4LL * B_ * DH_ + 2 * 256;
        int blocks = (int)((total + 255) / 256);
        prep_kernel<<<blocks, 256, 0, stream>>>(X, pad, WihF, WihB, WhhF, WhhB,
                                                xpad, wihF, wihB, whhF, whhB, hbuf, cnt);
    }
    // Phase 1: batched input projections (bf16 WMMA GEMM, both directions)
    {
        dim3 grid(64 * 12, 2);
        igemm_kernel<<<grid, 256, 0, stream>>>(xpad, wihF, wihB, bihF, bihB, giF, giB);
    }
    // Phase 2: recurrence (16 WGs/dir, 145,920 B dynamic LDS each)
    {
        (void)hipFuncSetAttribute((const void*)gru_step_kernel,
                                  hipFuncAttributeMaxDynamicSharedMemorySize, 145920);
        gru_step_kernel<<<32, 384, 145920, stream>>>(giF, giB, whhF, whhB,
                                                     bhhF, bhhB, hbuf, cnt, outh);
    }
    // Phase 3: decoder
    decode_kernel<<<32, 32, 0, stream>>>(outh, Wc, bc, emb, outc);
}
